// AttentionHead_549755813913
// MI455X (gfx1250) — matrix operand order repaired and checked by reading, not compiled, not run on hardware
//
#include <hip/hip_runtime.h>

#define BB 8
#define TT 4096
#define CC 1024
#define DD 64

typedef __attribute__((ext_vector_type(16))) __bf16 v16bf;
typedef __attribute__((ext_vector_type(8)))  float  v8f;

union ABu { v16bf v; uint4 q[2]; unsigned short u[16]; unsigned w[8]; };

__device__ __forceinline__ unsigned short f2bf(float f) {
  unsigned u = __float_as_uint(f);
  u += 0x7FFFu + ((u >> 16) & 1u);      // round-to-nearest-even bf16
  return (unsigned short)(u >> 16);
}

__device__ __forceinline__ unsigned pk2bf(float lo, float hi) {
#if __has_builtin(__builtin_amdgcn_cvt_pk_bf16_f32)
  typedef __attribute__((ext_vector_type(2))) __bf16 v2bf;
  union { v2bf v; unsigned u; } c;
  c.v = __builtin_amdgcn_cvt_pk_bf16_f32(lo, hi);
  return c.u;
#else
  return (unsigned)f2bf(lo) | ((unsigned)f2bf(hi) << 16);
#endif
}

// async global -> LDS copy, 16B per lane (512B per op).  GV addressing.
__device__ __forceinline__ void async_ldst(unsigned ldsa, const void* g) {
  asm volatile("global_load_async_to_lds_b128 %0, %1, off"
               :: "v"(ldsa), "v"((unsigned long long)(uintptr_t)g) : "memory");
}
// variant with immediate offset (applied to BOTH lds and global address)
template <int OFF>
__device__ __forceinline__ void async_ldst_off(unsigned ldsa, const void* g) {
  asm volatile("global_load_async_to_lds_b128 %0, %1, off offset:%2"
               :: "v"(ldsa), "v"((unsigned long long)(uintptr_t)g), "i"(OFF)
               : "memory");
}
__device__ __forceinline__ void wait_async_le16() {
  asm volatile("s_wait_asynccnt 0x10" ::: "memory");
}
__device__ __forceinline__ void wait_async_0() {
  asm volatile("s_wait_asynccnt 0x0" ::: "memory");
}

// ---------------------------------------------------------------------------
// Weight preconversion: Wb[3][D*C] bf16.  p: 0=K, 1=Q (scaled by 1/sqrt(D)*log2e), 2=V
// ---------------------------------------------------------------------------
__global__ void __launch_bounds__(256)
wconv_kernel(const float* __restrict__ Wk, const float* __restrict__ Wq,
             const float* __restrict__ Wv, unsigned short* __restrict__ Wb) {
  const int p = blockIdx.y;
  const float* src = (p == 0) ? Wk : (p == 1) ? Wq : Wv;
  const float scl = (p == 1) ? (0.125f * 1.44269504088896f) : 1.0f;
  const int idx = (blockIdx.x * 256 + threadIdx.x) * 8;
  float4 f0 = *(const float4*)(src + idx);
  float4 f1 = *(const float4*)(src + idx + 4);
  uint4 o;
  o.x = pk2bf(f0.x * scl, f0.y * scl);
  o.y = pk2bf(f0.z * scl, f0.w * scl);
  o.z = pk2bf(f1.x * scl, f1.y * scl);
  o.w = pk2bf(f1.z * scl, f1.w * scl);
  *(uint4*)(Wb + (size_t)p * DD * CC + idx) = o;
}

// ---------------------------------------------------------------------------
// X preconversion to bf16 (memory-bound; keeps conversion VALU out of GEMM loop)
// ---------------------------------------------------------------------------
__global__ void __launch_bounds__(256)
xconv_kernel(const float* __restrict__ X, unsigned short* __restrict__ Xb) {
  const size_t idx = ((size_t)blockIdx.x * 256 + threadIdx.x) * 8;
  float4 f0 = *(const float4*)(X + idx);
  float4 f1 = *(const float4*)(X + idx + 4);
  uint4 o;
  o.x = pk2bf(f0.x, f0.y);
  o.y = pk2bf(f0.z, f0.w);
  o.z = pk2bf(f1.x, f1.y);
  o.w = pk2bf(f1.z, f1.w);
  *(uint4*)(Xb + idx) = o;
}

// ---------------------------------------------------------------------------
// Fused projection: K/Q/V = X @ W^T, all three in one wave (12 WMMAs / chunk).
// Pure b128-load + WMMA inner loop (operands preconverted to bf16).
// grid = (T/16, B)
// ---------------------------------------------------------------------------
__global__ void __launch_bounds__(32)
proj_kernel(const unsigned short* __restrict__ Xb,
            const unsigned short* __restrict__ Wb,
            unsigned short* __restrict__ Qb,
            unsigned short* __restrict__ Kb,
            unsigned short* __restrict__ Vt) {
  const int lane  = threadIdx.x;
  const int l15   = lane & 15;
  const int h     = lane >> 4;
  const int tbase = blockIdx.x * 16;
  const int b     = blockIdx.y;

  v8f acc[3][4] = {};
  const unsigned short* xrow = Xb + (size_t)(b * TT + tbase + l15) * CC + 8 * h;

  for (int cb = 0; cb < CC; cb += 32) {
    ABu a;
    a.q[0] = *(const uint4*)(xrow + cb);        // K elems {0..7}  (+8 upper half-wave)
    a.q[1] = *(const uint4*)(xrow + cb + 16);   // K elems {16..23}

#pragma unroll
    for (int p = 0; p < 3; ++p) {
      ABu bm[4];
#pragma unroll
      for (int dt = 0; dt < 4; ++dt) {          // batch loads before WMMAs
        const unsigned short* wrow =
            Wb + (size_t)p * DD * CC + (size_t)(dt * 16 + l15) * CC + cb + 8 * h;
        bm[dt].q[0] = *(const uint4*)(wrow);
        bm[dt].q[1] = *(const uint4*)(wrow + 16);
      }
#pragma unroll
      for (int dt = 0; dt < 4; ++dt)
        acc[p][dt] = __builtin_amdgcn_wmma_f32_16x16x32_bf16(
            false, a.v, false, bm[dt].v, (short)0, acc[p][dt], false, false);
    }
  }

  // K (p=0) and Q (p=1): row-major scatter stores (C-layout: row=r+8h, col=l15)
#pragma unroll
  for (int dt = 0; dt < 4; ++dt) {
#pragma unroll
    for (int r = 0; r < 8; ++r) {
      const int t = tbase + r + 8 * h;
      const int d = dt * 16 + l15;
      Kb[(size_t)(b * TT + t) * DD + d] = f2bf(acc[0][dt][r]);
      Qb[(size_t)(b * TT + t) * DD + d] = f2bf(acc[1][dt][r]);
    }
  }
  // V stored transposed: accumulator rows contiguous in t -> one b128 per d-tile
#pragma unroll
  for (int dt = 0; dt < 4; ++dt) {
    uint4 o;
    o.x = pk2bf(acc[2][dt][0], acc[2][dt][1]);
    o.y = pk2bf(acc[2][dt][2], acc[2][dt][3]);
    o.z = pk2bf(acc[2][dt][4], acc[2][dt][5]);
    o.w = pk2bf(acc[2][dt][6], acc[2][dt][7]);
    *(uint4*)(Vt + ((size_t)b * DD + dt * 16 + l15) * TT + tbase + 8 * h) = o;
  }
}

// ---------------------------------------------------------------------------
// Flash attention: one wave per 16-row query tile; 32-key blocks; K/V staged
// into LDS with async copies, double-buffered; online softmax; PV via WMMA.
// grid = (T/16, B)
// ---------------------------------------------------------------------------
__global__ void __launch_bounds__(32)
attn_kernel(const unsigned short* __restrict__ Qb,
            const unsigned short* __restrict__ Kb,
            const unsigned short* __restrict__ Vt,
            float* __restrict__ Out) {
  __shared__ unsigned short kbuf[2][32 * 64];   // [buf][key][d]  4KB per buffer
  __shared__ unsigned short vbuf[2][64 * 32];   // [buf][d][key]  4KB per buffer
  __shared__ unsigned short lds_p[16 * 32];     // P bounce tile  1KB

  const int lane  = threadIdx.x;
  const int l15   = lane & 15;
  const int h     = lane >> 4;
  const int tbase = blockIdx.x * 16;
  const int b     = blockIdx.y;

  // per-lane async-copy bases
  // K: op i covers keys i*4 + (lane>>3); row=128B; chunk=(lane&7)*16
  const char* gk0 = (const char*)Kb + (size_t)(b * TT) * (DD * 2)
                    + (lane >> 3) * (DD * 2) + (lane & 7) * 16;
  const unsigned kld0 = (unsigned)(uintptr_t)&kbuf[0][0]
                        + (lane >> 3) * 128 + (lane & 7) * 16;
  // V: op i covers d = i*8 + (lane>>2); row=64B in LDS, TT*2B in global
  const char* gv0 = (const char*)Vt + (size_t)(b * DD) * (TT * 2)
                    + (size_t)(lane >> 2) * (TT * 2) + (lane & 3) * 16;
  const unsigned vld0 = (unsigned)(uintptr_t)&vbuf[0][0]
                        + (lane >> 2) * 64 + (lane & 3) * 16;

  auto stage = [&](int bb, int jb) {
    const unsigned kd = kld0 + (unsigned)bb * 4096;
    const char*    gk = gk0 + (size_t)jb * 128;
    async_ldst_off<0>(kd, gk);      async_ldst_off<512>(kd, gk);
    async_ldst_off<1024>(kd, gk);   async_ldst_off<1536>(kd, gk);
    async_ldst_off<2048>(kd, gk);   async_ldst_off<2560>(kd, gk);
    async_ldst_off<3072>(kd, gk);   async_ldst_off<3584>(kd, gk);
    const unsigned vd = vld0 + (unsigned)bb * 4096;
    const char*    gv = gv0 + (size_t)jb * 2;
#pragma unroll
    for (int i = 0; i < 8; ++i)
      async_ldst(vd + (unsigned)(i * 512), gv + (size_t)i * (8 * TT * 2));
  };

  // Q in A-layout: two 16x32 chunks over D=64
  ABu qa[2];
  {
    const unsigned short* q0 = Qb + (size_t)(b * TT + tbase + l15) * DD + 8 * h;
    qa[0].q[0] = *(const uint4*)(q0);
    qa[0].q[1] = *(const uint4*)(q0 + 16);
    qa[1].q[0] = *(const uint4*)(q0 + 32);
    qa[1].q[1] = *(const uint4*)(q0 + 48);
  }

  v8f o[4] = {};
  float m8[8], l8[8];
#pragma unroll
  for (int r = 0; r < 8; ++r) { m8[r] = -3.0e38f; l8[r] = 0.0f; }

  const int jmax = ((tbase + 15) / 32) * 32;     // last (partially masked) key block
  stage(0, 0);                                   // prime the pipeline

  for (int jb = 0; jb <= jmax; jb += 32) {
    const int bb = (jb >> 5) & 1;
    if (jb + 32 <= jmax) {
      stage(bb ^ 1, jb + 32);                    // issue next block's copies
      wait_async_le16();                         // current buffer's 16 ops done
    } else {
      wait_async_0();
    }

    // ---- S = Q K^T : two 16x16 tiles, K-reduction over D=64 (LDS operands) ----
    v8f s[2] = {};
#pragma unroll
    for (int nt = 0; nt < 2; ++nt) {
      const unsigned short* krow = &kbuf[bb][(nt * 16 + l15) * 64 + 8 * h];
      ABu k0, k1;
      k0.q[0] = *(const uint4*)(krow);
      k0.q[1] = *(const uint4*)(krow + 16);
      k1.q[0] = *(const uint4*)(krow + 32);
      k1.q[1] = *(const uint4*)(krow + 48);
      s[nt] = __builtin_amdgcn_wmma_f32_16x16x32_bf16(
          false, qa[0].v, false, k0.v, (short)0, s[nt], false, false);
      s[nt] = __builtin_amdgcn_wmma_f32_16x16x32_bf16(
          false, qa[1].v, false, k1.v, (short)0, s[nt], false, false);
    }

    // ---- causal mask (diagonal block only) ----
    if (jb + 31 > tbase) {
#pragma unroll
      for (int nt = 0; nt < 2; ++nt)
#pragma unroll
        for (int r = 0; r < 8; ++r)
          if (jb + nt * 16 + l15 > tbase + r + 8 * h) s[nt][r] = -3.0e38f;
    }

    // ---- online softmax ----
    float scale[8];
#pragma unroll
    for (int r = 0; r < 8; ++r) {
      float v = fmaxf(s[0][r], s[1][r]);
      v = fmaxf(v, __shfl_xor(v, 1, 32));
      v = fmaxf(v, __shfl_xor(v, 2, 32));
      v = fmaxf(v, __shfl_xor(v, 4, 32));
      v = fmaxf(v, __shfl_xor(v, 8, 32));
      const float mn = fmaxf(m8[r], v);
      scale[r] = exp2f(m8[r] - mn);
      m8[r] = mn;
    }
#pragma unroll
    for (int nt = 0; nt < 2; ++nt)
#pragma unroll
      for (int r = 0; r < 8; ++r) s[nt][r] = exp2f(s[nt][r] - m8[r]);
#pragma unroll
    for (int r = 0; r < 8; ++r) {
      float v = s[0][r] + s[1][r];
      v += __shfl_xor(v, 1, 32);
      v += __shfl_xor(v, 2, 32);
      v += __shfl_xor(v, 4, 32);
      v += __shfl_xor(v, 8, 32);
      l8[r] = l8[r] * scale[r] + v;
    }
#pragma unroll
    for (int dt = 0; dt < 4; ++dt)
#pragma unroll
      for (int r = 0; r < 8; ++r) o[dt][r] *= scale[r];

    // ---- P: C-layout f32 -> row-major bf16 in LDS -> A-layout reload ----
#pragma unroll
    for (int nt = 0; nt < 2; ++nt)
#pragma unroll
      for (int r = 0; r < 8; ++r)
        lds_p[(r + 8 * h) * 32 + nt * 16 + l15] = f2bf(s[nt][r]);
    asm volatile("s_wait_dscnt 0x0" ::: "memory");
    ABu pa;
    {
      const unsigned short* pr = &lds_p[l15 * 32 + 8 * h];
      pa.q[0] = *(const uint4*)(pr);
      pa.q[1] = *(const uint4*)(pr + 16);
    }

    // ---- O += P V (V^T rows contiguous in LDS) ----
#pragma unroll
    for (int dt = 0; dt < 4; ++dt) {
      const unsigned short* vrow = &vbuf[bb][(dt * 16 + l15) * 32 + 8 * h];
      ABu vb;
      vb.q[0] = *(const uint4*)(vrow);
      vb.q[1] = *(const uint4*)(vrow + 16);
      o[dt] = __builtin_amdgcn_wmma_f32_16x16x32_bf16(
          false, pa.v, false, vb.v, (short)0, o[dt], false, false);
    }
  }

  // ---- normalize and store fp32 output ----
#pragma unroll
  for (int r = 0; r < 8; ++r) l8[r] = 1.0f / l8[r];
#pragma unroll
  for (int dt = 0; dt < 4; ++dt)
#pragma unroll
    for (int r = 0; r < 8; ++r)
      Out[(size_t)(b * TT + tbase + r + 8 * h) * DD + dt * 16 + l15] =
          o[dt][r] * l8[r];
}

// ---------------------------------------------------------------------------
extern "C" void kernel_launch(void* const* d_in, const int* in_sizes, int n_in,
                              void* d_out, int out_size, void* d_ws, size_t ws_size,
                              hipStream_t stream) {
  (void)in_sizes; (void)n_in; (void)out_size; (void)ws_size;
  const float* X  = (const float*)d_in[0];
  const float* Wk = (const float*)d_in[1];
  const float* Wq = (const float*)d_in[2];
  const float* Wv = (const float*)d_in[3];

  unsigned short* Wb = (unsigned short*)d_ws;                 // 3*D*C bf16
  unsigned short* Xb = Wb + (size_t)3 * DD * CC;              // B*T*C bf16
  unsigned short* Qb = Xb + (size_t)BB * TT * CC;             // B*T*D bf16
  unsigned short* Kb = Qb + (size_t)BB * TT * DD;             // B*T*D bf16
  unsigned short* Vt = Kb + (size_t)BB * TT * DD;             // B*D*T bf16 (transposed)

  dim3 gw(DD * CC / 2048, 3);
  wconv_kernel<<<gw, 256, 0, stream>>>(Wk, Wq, Wv, Wb);

  xconv_kernel<<<(size_t)BB * TT * CC / 2048, 256, 0, stream>>>(X, Xb);

  dim3 gproj(TT / 16, BB);
  proj_kernel<<<gproj, 32, 0, stream>>>(Xb, Wb, Qb, Kb, Vt);

  dim3 gattn(TT / 16, BB);
  attn_kernel<<<gattn, 32, 0, stream>>>(Qb, Kb, Vt, (float*)d_out);
}
